// minerva_wrapper_37211596652568
// MI455X (gfx1250) — matrix operand order repaired and checked by reading, not compiled, not run on hardware
//
#include <hip/hip_runtime.h>
#include <math.h>

// ---------------------------------------------------------------------------
// Minerva fused pipeline for MI455X (gfx1250, wave32, WMMA f16->f32)
//
//   Xg = X @ g_w^T + g_b           [8192, 512]   (WMMA, f16 out)
//   Dg = D @ g_w^T + g_b           [16384, 512]  (WMMA, f16 out)
//   row-L2-normalize Xg, Dg in place
//   echo[i] = sum_j (X̂g[i]·D̂g[j])^3 * (2 r[j]-1)   (WMMA + fused epilogue,
//                                                    a[8192,16384] NEVER materialized)
//   out = sigmoid(echo * h_w + h_b)
//
// Big-GEMM B tiles are staged global->LDS once per block (8x traffic cut),
// via CDNA5 async-to-LDS ops when the builtin is available.
// ---------------------------------------------------------------------------

typedef __attribute__((ext_vector_type(16))) _Float16 v16h;
typedef __attribute__((ext_vector_type(8)))  _Float16 v8h;
typedef __attribute__((ext_vector_type(8)))  float    v8f;
typedef __attribute__((ext_vector_type(4)))  int      v4i;

#define N_Q   8192
#define M_EX  16384
#define F_IN  768
#define E_DIM 512
#define KSTEPS (E_DIM / 32)   // 16

#if __has_builtin(__builtin_amdgcn_global_load_async_to_lds_b128) && \
    __has_builtin(__builtin_amdgcn_s_wait_asynccnt)
#define USE_ASYNC_LDS 1
typedef __attribute__((address_space(1))) v4i* gv4i_ptr;   // global int4*
typedef __attribute__((address_space(3))) v4i* lv4i_ptr;   // LDS int4*
#else
#define USE_ASYNC_LDS 0
#endif

// ---------------------------------------------------------------------------
// fp32 -> fp16 conversion (for g_w)
// ---------------------------------------------------------------------------
__global__ __launch_bounds__(256)
void cvt_f32_f16_kernel(const float* __restrict__ src, _Float16* __restrict__ dst, int n)
{
    int i = blockIdx.x * 256 + threadIdx.x;
    if (i < n) dst[i] = (_Float16)src[i];
}

// ---------------------------------------------------------------------------
// Embedding GEMM: dst[row, e] = sum_k src[row,k] * gwh[e,k] + gb[e]
//   src: f32 [rows, 768] row-major; gwh: f16 [512, 768] row-major
//   grid: (rows/128, 512/64), block: 256 (8 waves); each wave: 16 rows x 64 cols
// ---------------------------------------------------------------------------
__global__ __launch_bounds__(256)
void embed_kernel(const float* __restrict__ src, const _Float16* __restrict__ gwh,
                  const float* __restrict__ gb, _Float16* __restrict__ dst)
{
    const int tid    = threadIdx.x;
    const int lane   = tid & 31;
    const int l16    = lane & 15;
    const int kg     = lane >> 4;          // k-group selector AND C-half selector
    const int waveId = tid >> 5;
    const int rowBase = blockIdx.x * 128 + waveId * 16;
    const int colBase = blockIdx.y * 64;

    const float* srcRow = src + (size_t)(rowBase + l16) * F_IN;

    v8f c[4];
    #pragma unroll
    for (int t = 0; t < 4; ++t) c[t] = (v8f){0.f,0.f,0.f,0.f,0.f,0.f,0.f,0.f};

    for (int kb = 0; kb < F_IN; kb += 32) {
        // A fragment: 16x32 f16 (convert from f32 on the fly)
        v16h a;
        #pragma unroll
        for (int i = 0; i < 8; ++i) {
            a[i]     = (_Float16)srcRow[kb + kg * 16 + i];
            a[8 + i] = (_Float16)srcRow[kb + kg * 16 + 8 + i];
        }
        // B fragments: B[k, n] = gwh[colBase + t*16 + n, k] -> contiguous 32B loads
        const int koff = kb + kg * 16;
        v16h b[4];
        #pragma unroll
        for (int t = 0; t < 4; ++t)
            b[t] = *(const v16h*)(gwh + (size_t)(colBase + t * 16 + l16) * F_IN + koff);
        #pragma unroll
        for (int t = 0; t < 4; ++t)
            c[t] = __builtin_amdgcn_wmma_f32_16x16x32_f16(
                       false, a, false, b[t], (short)0, c[t], false, false);
    }

    // Epilogue: add bias, store f16 (normalization is a later pass)
    #pragma unroll
    for (int t = 0; t < 4; ++t) {
        const int colg = colBase + t * 16 + l16;
        const float bias = gb[colg];
        #pragma unroll
        for (int rr = 0; rr < 8; ++rr) {
            const int rowM = rowBase + rr + (kg ? 8 : 0);
            dst[(size_t)rowM * E_DIM + colg] = (_Float16)(c[t][rr] + bias);
        }
    }
}

// ---------------------------------------------------------------------------
// In-place row L2-normalization of a f16 [rows, 512] buffer. One block per row.
// ---------------------------------------------------------------------------
__global__ __launch_bounds__(256)
void normalize_kernel(_Float16* __restrict__ buf)
{
    _Float16* p = buf + (size_t)blockIdx.x * E_DIM;
    const int tid = threadIdx.x;
    const float x0 = (float)p[tid * 2];
    const float x1 = (float)p[tid * 2 + 1];

    __shared__ float red[256];
    red[tid] = x0 * x0 + x1 * x1;
    __syncthreads();
    #pragma unroll
    for (int off = 128; off > 0; off >>= 1) {
        if (tid < off) red[tid] += red[tid + off];
        __syncthreads();
    }
    const float scale = 1.0f / fmaxf(sqrtf(red[0]), 1e-12f);
    p[tid * 2]     = (_Float16)(x0 * scale);
    p[tid * 2 + 1] = (_Float16)(x1 * scale);
}

// ---------------------------------------------------------------------------
// zero init for echo accumulator
// ---------------------------------------------------------------------------
__global__ __launch_bounds__(256)
void zero_kernel(float* __restrict__ p, int n)
{
    int i = blockIdx.x * 256 + threadIdx.x;
    if (i < n) p[i] = 0.f;
}

// ---------------------------------------------------------------------------
// Fused similarity kernel: never materializes a[8192,16384].
//   echo[i] += sum over this block's columns of (Xg[i]·Dg[j])^3 * (2 r[j] - 1)
//
// grid: (8192/128, 16384/512); block 256 (8 waves); each wave: 16 rows x 64 cols
// per chunk. The 64-col x 512-K B tile (64 KB) is staged in LDS ONCE per block
// per chunk in *fragment order* [kstep][tile][lane][16 halves], so every wave's
// B operand is a lane-consecutive 32B LDS read (conflict-free) and global B
// traffic drops 8x (one read per block instead of per wave).
// ---------------------------------------------------------------------------
__global__ __launch_bounds__(256)
void minerva_fused_kernel(const _Float16* __restrict__ Xg,
                          const _Float16* __restrict__ Dg,
                          const float* __restrict__ rlab,
                          float* __restrict__ echo)
{
    __shared__ _Float16 sB[KSTEPS * 4 * 32 * 16];   // 32768 halves = 64 KB

    const int tid    = threadIdx.x;
    const int lane   = tid & 31;
    const int l16    = lane & 15;
    const int kg     = lane >> 4;
    const int waveId = tid >> 5;
    const int rowBase  = blockIdx.x * 128 + waveId * 16;
    const int colSuper = blockIdx.y * 512;

    const _Float16* xrow = Xg + (size_t)(rowBase + l16) * E_DIM;

    float eacc[8];
    #pragma unroll
    for (int rr = 0; rr < 8; ++rr) eacc[rr] = 0.f;

    for (int chunk = 0; chunk < 8; ++chunk) {
        const int colBase = colSuper + chunk * 64;

        // ---- stage B tile: 4096 x 16B pieces, fragment-ordered ----
        __syncthreads();   // previous chunk's LDS reads must complete
        #pragma unroll
        for (int j = 0; j < 16; ++j) {
            const int c    = j * 256 + tid;        // 0..4095 (16B pieces)
            const int h    = c & 1;                // low/high half of a fragment
            const int fi   = c >> 1;               // fragment 0..2047
            const int fln  = fi & 31;
            const int ft   = (fi >> 5) & 3;
            const int fks  = fi >> 7;
            const int col  = colBase + ft * 16 + (fln & 15);
            const int koff = fks * 32 + (fln >> 4) * 16 + h * 8;
            const _Float16* gsrc = Dg + (size_t)col * E_DIM + koff;
            _Float16* ldst = sB + (size_t)c * 8;
#if USE_ASYNC_LDS
            __builtin_amdgcn_global_load_async_to_lds_b128(
                (gv4i_ptr)gsrc, (lv4i_ptr)ldst, 0, 0);
#else
            *(v8h*)ldst = *(const v8h*)gsrc;
#endif
        }
#if USE_ASYNC_LDS
        __builtin_amdgcn_s_wait_asynccnt(0);
#endif
        __syncthreads();

        // ---- 16x64 per wave: 4 accumulator tiles, K = 512 ----
        v8f c4[4];
        #pragma unroll
        for (int t = 0; t < 4; ++t) c4[t] = (v8f){0.f,0.f,0.f,0.f,0.f,0.f,0.f,0.f};

        for (int kb = 0; kb < E_DIM; kb += 32) {
            // A fragment: two contiguous 16B global loads per lane (WGP$/L2 hot)
            const v8h lo = *(const v8h*)(xrow + kb + kg * 16);
            const v8h hi = *(const v8h*)(xrow + kb + kg * 16 + 8);
            v16h a;
            #pragma unroll
            for (int i = 0; i < 8; ++i) { a[i] = lo[i]; a[8 + i] = hi[i]; }

            const int kstep = kb >> 5;
            #pragma unroll
            for (int t = 0; t < 4; ++t) {
                const v16h b = *(const v16h*)(sB + (((kstep * 4 + t) * 32 + lane) << 4));
                c4[t] = __builtin_amdgcn_wmma_f32_16x16x32_f16(
                            false, a, false, b, (short)0, c4[t], false, false);
            }
        }

        // Fused epilogue: a^3 * (2r-1), accumulate per-lane column partials.
        // C layout: lane l, vgpr rr holds (M = rr + (l>=16 ? 8 : 0), N = l&15)
        #pragma unroll
        for (int t = 0; t < 4; ++t) {
            const int colg = colBase + t * 16 + l16;
            const float rv = rlab[colg] * 2.0f - 1.0f;
            #pragma unroll
            for (int rr = 0; rr < 8; ++rr) {
                const float v = c4[t][rr];
                eacc[rr] += v * v * v * rv;   // |a|^3 * sign(a) == a^3
            }
        }
    }

    // Cross-lane reduce over N: xor masks 1,2,4,8 stay inside each 16-lane half,
    // matching the M-split (lanes 0-15 -> rows rr, lanes 16-31 -> rows rr+8).
    #pragma unroll
    for (int rr = 0; rr < 8; ++rr) {
        float v = eacc[rr];
        v += __shfl_xor(v, 1);
        v += __shfl_xor(v, 2);
        v += __shfl_xor(v, 4);
        v += __shfl_xor(v, 8);
        if (l16 == 0) {
            const int rowM = rowBase + rr + (kg ? 8 : 0);
            atomicAdd(&echo[rowM], v);   // global_atomic_add_f32, L2-resident
        }
    }
}

// ---------------------------------------------------------------------------
// final: out = sigmoid(echo * h_w + h_b)
// ---------------------------------------------------------------------------
__global__ __launch_bounds__(256)
void final_kernel(const float* __restrict__ echo, const float* __restrict__ hw,
                  const float* __restrict__ hb, float* __restrict__ out, int n)
{
    int i = blockIdx.x * 256 + threadIdx.x;
    if (i < n) {
        const float z = echo[i] * hw[0] + hb[0];
        out[i] = 1.0f / (1.0f + __expf(-z));
    }
}

// ---------------------------------------------------------------------------
extern "C" void kernel_launch(void* const* d_in, const int* in_sizes, int n_in,
                              void* d_out, int out_size, void* d_ws, size_t ws_size,
                              hipStream_t stream)
{
    const float* X  = (const float*)d_in[0];   // [8192, 768]
    const float* D  = (const float*)d_in[1];   // [16384, 768]
    const float* r  = (const float*)d_in[2];   // [16384, 1]
    const float* gw = (const float*)d_in[3];   // [512, 768]
    const float* gb = (const float*)d_in[4];   // [512]
    const float* hw = (const float*)d_in[5];   // [1,1]
    const float* hb = (const float*)d_in[6];   // [1]
    float* out = (float*)d_out;                // [8192, 1] f32

    // workspace layout (all offsets 256B aligned), total ~24.8 MB
    char* ws = (char*)d_ws;
    _Float16* gwh  = (_Float16*)(ws);                               // 512*768*2   = 786432
    _Float16* Xg   = (_Float16*)(ws + 786432);                      // 8192*512*2  = 8388608
    _Float16* Dg   = (_Float16*)(ws + 786432 + 8388608);            // 16384*512*2 = 16777216
    float*    echo = (float*)   (ws + 786432 + 8388608 + 16777216); // 8192*4

    // 1) convert g_w to f16 once per call
    cvt_f32_f16_kernel<<<(E_DIM * F_IN) / 256, 256, 0, stream>>>(gw, gwh, E_DIM * F_IN);

    // 2) embedding GEMMs (WMMA)
    embed_kernel<<<dim3(N_Q  / 128, E_DIM / 64), 256, 0, stream>>>(X, gwh, gb, Xg);
    embed_kernel<<<dim3(M_EX / 128, E_DIM / 64), 256, 0, stream>>>(D, gwh, gb, Dg);

    // 3) row L2-normalize in place
    normalize_kernel<<<N_Q,  256, 0, stream>>>(Xg);
    normalize_kernel<<<M_EX, 256, 0, stream>>>(Dg);

    // 4) fused similarity + power-sign + label-weighted reduction
    zero_kernel<<<N_Q / 256, 256, 0, stream>>>(echo, N_Q);
    minerva_fused_kernel<<<dim3(N_Q / 128, M_EX / 512), 256, 0, stream>>>(Xg, Dg, r, echo);

    // 5) sigmoid head
    final_kernel<<<N_Q / 256, 256, 0, stream>>>(echo, hw, hb, out, N_Q);
}
